// PerformerAttention_72335839200085
// MI455X (gfx1250) — compile-verified, matrix operand order checked by reading, not run on hardware
//
#include <hip/hip_runtime.h>
#include <hip/hip_bf16.h>

// ---------------------------------------------------------------------------
// Performer attention for MI455X (gfx1250): wave32, WMMA f16->f32,
// async global->LDS staging (ASYNCcnt), fragment-native LDS swizzle.
// ---------------------------------------------------------------------------

typedef __attribute__((ext_vector_type(16))) _Float16 v16h;
typedef __attribute__((ext_vector_type(8)))  _Float16 v8h;
typedef __attribute__((ext_vector_type(4)))  _Float16 v4h;
typedef __attribute__((ext_vector_type(8)))  float    v8f;

constexpr int Bc  = 4;
constexpr int Sc  = 8192;
constexpr int Ec  = 768;
constexpr int Hc  = 12;
constexpr int Dh  = 64;
constexpr int Mc  = Bc * Sc;          // 32768
constexpr int NSPLIT = 32;

// GEMM tiling: 128x128 block tile, K-step 32, double-buffered LDS (32 KB).
constexpr int BM  = 128;
constexpr int BN  = 128;
constexpr int BK  = 32;
constexpr int BKP = 34;               // padded stride for the small kernels

// ---------------------------------------------------------------------------
// CDNA5 async copy: 16B global -> LDS, tracked by ASYNCcnt (ISA 10.x / 15.18).
// ldsByteOff = low 32 bits of a flat __shared__ pointer (LDS aperture truncates
// to addr[31:0], so this is the wave-relative LDS byte address).
// ---------------------------------------------------------------------------
__device__ __forceinline__ void asyncLds16(unsigned ldsByteOff, const void* gaddr) {
  asm volatile("global_load_async_to_lds_b128 %0, %1, off"
               : : "v"(ldsByteOff), "v"(gaddr) : "memory");
}
__device__ __forceinline__ void waitAsync0() {
  asm volatile("s_wait_asynccnt 0x0" ::: "memory");
}

// ---------------------------------------------------------------------------
// Fragment-native LDS layout (per 128-row x 32-K f16 tile, 8 KB):
//   record(row,kb) : 8 consecutive f16 (16 B), kb in 0..3 -> K = kb*8..kb*8+7
//   plane = kb>>1, g = kb&1
//   rec = plane*256 + (row>>4)*32 + g*16 + (row&15)
// Fragment read (ISA 7.12.2, lane = g*16+r): elems 0..7 from plane0 record,
// elems 8..15 from plane1 record -> two ds_load_b128, 16B lane stride.
// ---------------------------------------------------------------------------
__device__ __forceinline__ v16h fragN(const _Float16* buf, int rowGroup, int lane) {
  const int g = (lane >> 4) & 1;
  const int r = lane & 15;
  const int rec0 = rowGroup * 32 + g * 16 + r;          // plane 0
  v8h lo = *(const v8h*)(buf + rec0 * 8);
  v8h hi = *(const v8h*)(buf + (rec0 + 256) * 8);       // plane 1
  return __builtin_shufflevector(lo, hi, 0,1,2,3,4,5,6,7,8,9,10,11,12,13,14,15);
}

// Generic padded-row fragment (used by the small per-head kernels).
__device__ __forceinline__ v16h ldsFrag(const _Float16* base, int row0,
                                        int ldStride, int k0, int lane) {
  const int g = (lane >> 4) & 1;
  const int r = lane & 15;
  const _Float16* p = base + (row0 + r) * ldStride + k0 + g * 8;
  v16h f;
#pragma unroll
  for (int j = 0; j < 8; ++j) { f[j] = p[j]; f[j + 8] = p[16 + j]; }
  return f;
}

__device__ __forceinline__ v8f wmma16(v16h a, v16h b, v8f c) {
  return __builtin_amdgcn_wmma_f32_16x16x32_f16(false, a, false, b,
                                                (short)0, c, false, false);
}

// ---------------------------------------------------------------------------
// f32 -> f16 pre-conversion (activations + weights), vectorized.
// ---------------------------------------------------------------------------
__global__ __launch_bounds__(256)
void cvt_f32_f16_kernel(const float* __restrict__ in, _Float16* __restrict__ out,
                        int n) {
  const int i = (blockIdx.x * 256 + threadIdx.x) * 4;
  if (i >= n) return;
  float4 v = *(const float4*)(in + i);
  v4h h;
  h[0] = (_Float16)v.x; h[1] = (_Float16)v.y;
  h[2] = (_Float16)v.z; h[3] = (_Float16)v.w;
  *(v4h*)(out + i) = h;
}

// ---------------------------------------------------------------------------
// GEMM: out[m,n] = act( sum_k A[m,k]*W[n,k] + bias[n] ), A/W f16, acc f32.
// 256 threads = 8 waves; wave (wm 0..3, wn 0..1) owns 32x64 = 2x4 WMMA frags.
// Double-buffered async staging: issue next tile, compute current, wait+barrier.
// ---------------------------------------------------------------------------
__global__ __launch_bounds__(256)
void gemm_f16_kernel(const _Float16* __restrict__ A,
                     const _Float16* __restrict__ W,
                     const float* __restrict__ bias,
                     void* __restrict__ outP,
                     int Mtot, int N, int K, int doRelu, int outHalf) {
  // [buf(2)][ A(4096 f16) | B(4096 f16) ]  = 32 KB
  __shared__ _Float16 smem[2 * 8192];
  const unsigned smemBase = (unsigned)(size_t)(void*)smem;

  const int tid  = threadIdx.x;
  const int lane = tid & 31;
  const int wid  = tid >> 5;
  const int wm   = wid >> 1;           // M offset wm*32
  const int wn   = wid & 1;            // N offset wn*64
  const int n0   = blockIdx.x * BN;
  const int m0   = blockIdx.y * BM;

  // Per-thread copy records: 2 per matrix (512 records / 256 threads).
  unsigned ldsRec[2];
  const _Float16* gA[2];
  const _Float16* gB[2];
#pragma unroll
  for (int j = 0; j < 2; ++j) {
    const int ri    = tid + j * 256;
    const int row   = ri >> 2;
    const int kb    = ri & 3;
    const int plane = (kb >> 1) & 1;
    const int g     = kb & 1;
    const int rec   = plane * 256 + ((row >> 4) * 32 + g * 16 + (row & 15));
    ldsRec[j] = (unsigned)(rec * 16);
    gA[j] = A + (size_t)(m0 + row) * K + kb * 8;
    gB[j] = W + (size_t)(n0 + row) * K + kb * 8;
  }

  auto issueTile = [&](int par) {
    const unsigned base = smemBase + (unsigned)par * 16384u;   // buf stride 16 KB
#pragma unroll
    for (int j = 0; j < 2; ++j) {
      asyncLds16(base + ldsRec[j],          gA[j]);
      asyncLds16(base + 8192u + ldsRec[j],  gB[j]);            // B region +8 KB
      gA[j] += BK;
      gB[j] += BK;
    }
  };

  v8f acc[2][4];
#pragma unroll
  for (int mi = 0; mi < 2; ++mi)
#pragma unroll
    for (int ni = 0; ni < 4; ++ni)
#pragma unroll
      for (int j = 0; j < 8; ++j) acc[mi][ni][j] = 0.0f;

  const int KSTEPS = K / BK;   // 24
  issueTile(0);
  waitAsync0();
  __syncthreads();

  for (int it = 0; it < KSTEPS; ++it) {
    const int par = it & 1;
    if (it + 1 < KSTEPS) issueTile((it + 1) & 1);

    const _Float16* bufA = smem + par * 8192;
    const _Float16* bufB = bufA + 4096;
    v16h a0 = fragN(bufA, wm * 2 + 0, lane);
    v16h a1 = fragN(bufA, wm * 2 + 1, lane);
#pragma unroll
    for (int ni = 0; ni < 4; ++ni) {
      v16h bb = fragN(bufB, wn * 4 + ni, lane);
      acc[0][ni] = wmma16(a0, bb, acc[0][ni]);
      acc[1][ni] = wmma16(a1, bb, acc[1][ni]);
    }

    waitAsync0();
    __syncthreads();
  }

  // Epilogue: bias + optional ReLU; f16 (intermediate) or f32 (final) store.
  float*    outF = (float*)outP;
  _Float16* outH = (_Float16*)outP;
  const int g = (lane >> 4) & 1;
  const int r = lane & 15;
#pragma unroll
  for (int ni = 0; ni < 4; ++ni) {
    const int n  = n0 + wn * 64 + ni * 16 + r;
    const float bv = bias[n];
#pragma unroll
    for (int mi = 0; mi < 2; ++mi) {
#pragma unroll
      for (int i = 0; i < 8; ++i) {
        const int m = m0 + wm * 32 + mi * 16 + g * 8 + i;
        float v = acc[mi][ni][i] + bv;
        if (doRelu) v = fmaxf(v, 0.0f);
        if (outHalf) outH[(size_t)m * N + n] = (_Float16)v;
        else         outF[(size_t)m * N + n] = v;
      }
    }
  }
}

// ---------------------------------------------------------------------------
// KV partials: part[d,e] = sum_{l in chunk} K[b,l,h,d]*V[b,l,h,e]; f16 inputs.
// 128 threads = 4 waves; wave w owns d-rows [16w,16w+16). Deterministic.
// ---------------------------------------------------------------------------
__global__ __launch_bounds__(128)
void kv_partial_kernel(const _Float16* __restrict__ Kf,
                       const _Float16* __restrict__ Vf,
                       float* __restrict__ part) {
  __shared__ _Float16 lK[Dh * BKP];   // [d][l]
  __shared__ _Float16 lV[Dh * BKP];   // [e][l]

  const int tid  = threadIdx.x;
  const int lane = tid & 31;
  const int w    = tid >> 5;
  const int bh   = blockIdx.x;
  const int spl  = blockIdx.y;
  const int b    = bh / Hc;
  const int h    = bh % Hc;
  const int chunk = Sc / NSPLIT;      // 256

  v8f acc[4];
#pragma unroll
  for (int ni = 0; ni < 4; ++ni)
#pragma unroll
    for (int j = 0; j < 8; ++j) acc[ni][j] = 0.0f;

  for (int it = 0; it < chunk / BK; ++it) {   // 8 K-steps of 32
    const int l0 = spl * chunk + it * BK;
    const _Float16* Kb = Kf + ((size_t)b * Sc + l0) * Ec + h * Dh;
    const _Float16* Vb = Vf + ((size_t)b * Sc + l0) * Ec + h * Dh;

    // 32 rows(l) x 64 (head dim) -> transposed LDS. 256 v8h recs / 128 thr.
#pragma unroll
    for (int i = tid; i < BK * Dh / 8; i += 128) {
      const int l  = i >> 3;
      const int c8 = (i & 7) << 3;
      v8h vk = *(const v8h*)(Kb + (size_t)l * Ec + c8);
      v8h vv = *(const v8h*)(Vb + (size_t)l * Ec + c8);
#pragma unroll
      for (int j = 0; j < 8; ++j) {
        lK[(c8 + j) * BKP + l] = vk[j];
        lV[(c8 + j) * BKP + l] = vv[j];
      }
    }
    __syncthreads();

    v16h a = ldsFrag(lK, w * 16, BKP, 0, lane);
#pragma unroll
    for (int ni = 0; ni < 4; ++ni) {
      v16h bfr = ldsFrag(lV, ni * 16, BKP, 0, lane);
      acc[ni] = wmma16(a, bfr, acc[ni]);
    }
    __syncthreads();
  }

  const int g = (lane >> 4) & 1;
  const int r = lane & 15;
  float* pb = part + ((size_t)bh * NSPLIT + spl) * (Dh * Dh);
#pragma unroll
  for (int ni = 0; ni < 4; ++ni) {
    const int e = ni * 16 + r;
#pragma unroll
    for (int i = 0; i < 8; ++i) {
      const int d = w * 16 + g * 8 + i;
      pb[d * Dh + e] = acc[ni][i];
    }
  }
}

__global__ __launch_bounds__(256)
void kv_reduce_kernel(const float* __restrict__ part, float* __restrict__ KV) {
  const int idx = blockIdx.x * blockDim.x + threadIdx.x;
  if (idx >= Bc * Hc * Dh * Dh) return;
  const int bh = idx / (Dh * Dh);
  const int de = idx % (Dh * Dh);
  float s = 0.0f;
  const float* p = part + (size_t)bh * NSPLIT * (Dh * Dh) + de;
#pragma unroll
  for (int k = 0; k < NSPLIT; ++k) s += p[(size_t)k * (Dh * Dh)];
  KV[idx] = s;
}

// ---------------------------------------------------------------------------
// QKV[b,l,h,e] = sum_d Q[b,l,h,d] * KV[b,h][d,e]; Q f16, KV f32, out f16.
// ---------------------------------------------------------------------------
__global__ __launch_bounds__(256)
void q_kv_kernel(const _Float16* __restrict__ Qf,
                 const float* __restrict__ KV,
                 _Float16* __restrict__ QKV) {
  constexpr int LDQ = 72;              // 144 B rows: 16B-aligned for b128 stores
  __shared__ _Float16 lQ[128 * LDQ];
  __shared__ _Float16 lKV[Dh * LDQ];   // [e][d]

  const int tid  = threadIdx.x;
  const int lane = tid & 31;
  const int w    = tid >> 5;
  const int bh   = blockIdx.x;
  const int tl   = blockIdx.y * 128;
  const int b    = bh / Hc;
  const int h    = bh % Hc;

  const float* KVb = KV + (size_t)bh * (Dh * Dh);
#pragma unroll
  for (int i = tid; i < Dh * Dh / 4; i += 256) {
    const int d  = i >> 4;
    const int e4 = (i & 15) << 2;
    float4 v = *(const float4*)(KVb + d * Dh + e4);
    lKV[(e4 + 0) * LDQ + d] = (_Float16)v.x;
    lKV[(e4 + 1) * LDQ + d] = (_Float16)v.y;
    lKV[(e4 + 2) * LDQ + d] = (_Float16)v.z;
    lKV[(e4 + 3) * LDQ + d] = (_Float16)v.w;
  }
  const _Float16* Qb = Qf + ((size_t)b * Sc + tl) * Ec + h * Dh;
#pragma unroll
  for (int i = tid; i < 128 * Dh / 8; i += 256) {
    const int l  = i >> 3;
    const int d8 = (i & 7) << 3;
    v8h v = *(const v8h*)(Qb + (size_t)l * Ec + d8);
    *(v8h*)(lQ + l * LDQ + d8) = v;    // contiguous -> ds_store_b128
  }
  __syncthreads();

  v8f acc[4];
#pragma unroll
  for (int ni = 0; ni < 4; ++ni)
#pragma unroll
    for (int j = 0; j < 8; ++j) acc[ni][j] = 0.0f;

#pragma unroll
  for (int ks = 0; ks < Dh; ks += BK) {
    v16h a = ldsFrag(lQ, w * 16, LDQ, ks, lane);
#pragma unroll
    for (int ni = 0; ni < 4; ++ni) {
      v16h bfr = ldsFrag(lKV, ni * 16, LDQ, ks, lane);
      acc[ni] = wmma16(a, bfr, acc[ni]);
    }
  }

  const int g = (lane >> 4) & 1;
  const int r = lane & 15;
#pragma unroll
  for (int ni = 0; ni < 4; ++ni) {
    const int e = ni * 16 + r;
#pragma unroll
    for (int i = 0; i < 8; ++i) {
      const int l = tl + w * 16 + g * 8 + i;
      QKV[((size_t)b * Sc + l) * Ec + h * Dh + e] = (_Float16)acc[ni][i];
    }
  }
}

// ---------------------------------------------------------------------------
extern "C" void kernel_launch(void* const* d_in, const int* in_sizes, int n_in,
                              void* d_out, int out_size, void* d_ws, size_t ws_size,
                              hipStream_t stream) {
  const float* query = (const float*)d_in[0];
  const float* key   = (const float*)d_in[1];
  const float* value = (const float*)d_in[2];
  const float* Wq    = (const float*)d_in[3];
  const float* bq    = (const float*)d_in[4];
  const float* Wk    = (const float*)d_in[5];
  const float* bk    = (const float*)d_in[6];
  const float* Wv    = (const float*)d_in[7];
  const float* bv    = (const float*)d_in[8];
  const float* Wo    = (const float*)d_in[9];
  const float* bo    = (const float*)d_in[10];

  const size_t AS  = (size_t)Mc * Ec;   // 25,165,824 elems
  const size_t WSZ = (size_t)Ec * Ec;   // 589,824 elems

  // Workspace carve-up (f16 regions first, then f32). QKVh aliases Kp.
  _Float16* qh  = (_Float16*)d_ws;
  _Float16* kh  = qh  + AS;
  _Float16* vh  = kh  + AS;
  _Float16* Wqh = vh  + AS;
  _Float16* Wkh = Wqh + WSZ;
  _Float16* Wvh = Wkh + WSZ;
  _Float16* Woh = Wvh + WSZ;
  _Float16* Qp  = Woh + WSZ;
  _Float16* Kp  = Qp  + AS;
  _Float16* Vp  = Kp  + AS;
  float*    part = (float*)(Vp + AS);
  float*    KVt  = part + (size_t)Bc * Hc * NSPLIT * Dh * Dh;
  _Float16* QKVh = Kp;                  // alias: K consumed before q_kv

  // 1) Pre-convert activations + weights to f16.
  cvt_f32_f16_kernel<<<(int)(AS / 1024), 256, 0, stream>>>(query, qh, (int)AS);
  cvt_f32_f16_kernel<<<(int)(AS / 1024), 256, 0, stream>>>(key,   kh, (int)AS);
  cvt_f32_f16_kernel<<<(int)(AS / 1024), 256, 0, stream>>>(value, vh, (int)AS);
  cvt_f32_f16_kernel<<<(int)(WSZ / 1024), 256, 0, stream>>>(Wq, Wqh, (int)WSZ);
  cvt_f32_f16_kernel<<<(int)(WSZ / 1024), 256, 0, stream>>>(Wk, Wkh, (int)WSZ);
  cvt_f32_f16_kernel<<<(int)(WSZ / 1024), 256, 0, stream>>>(Wv, Wvh, (int)WSZ);
  cvt_f32_f16_kernel<<<(int)(WSZ / 1024), 256, 0, stream>>>(Wo, Woh, (int)WSZ);

  const dim3 gemmGrid(Ec / BN, Mc / BM);   // (6, 256)

  // 2) Projections (Q,K with ReLU feature map), f16 outputs.
  gemm_f16_kernel<<<gemmGrid, 256, 0, stream>>>(qh, Wqh, bq, Qp, Mc, Ec, Ec, 1, 1);
  gemm_f16_kernel<<<gemmGrid, 256, 0, stream>>>(kh, Wkh, bk, Kp, Mc, Ec, Ec, 1, 1);
  gemm_f16_kernel<<<gemmGrid, 256, 0, stream>>>(vh, Wvh, bv, Vp, Mc, Ec, Ec, 0, 1);

  // 3) KV = K^T @ V per (b,h): split-S partials + deterministic reduce.
  kv_partial_kernel<<<dim3(Bc * Hc, NSPLIT), 128, 0, stream>>>(Kp, Vp, part);
  kv_reduce_kernel<<<(Bc * Hc * Dh * Dh + 255) / 256, 256, 0, stream>>>(part, KVt);

  // 4) QKV = Q @ KV per (b,h), f16 out.
  q_kv_kernel<<<dim3(Bc * Hc, Sc / 128), 256, 0, stream>>>(Qp, KVt, QKVh);

  // 5) Output projection, f32 to d_out.
  gemm_f16_kernel<<<gemmGrid, 256, 0, stream>>>(QKVh, Woh, bo, d_out, Mc, Ec, Ec, 0, 0);
}
